// ScoreNetwork_8340826488879
// MI455X (gfx1250) — matrix operand order repaired and checked by reading, not compiled, not run on hardware
//
#include <hip/hip_runtime.h>

typedef __attribute__((ext_vector_type(16))) __bf16 v16bf;
typedef __attribute__((ext_vector_type(8)))  float  v8f;
typedef __attribute__((ext_vector_type(4)))  unsigned int v4u;

union FragBF { v4u u[2]; v16bf v; };

__device__ __forceinline__ unsigned short f2bf(float x){
    unsigned u = __float_as_uint(x);
    u += 0x7FFFu + ((u >> 16) & 1u);          // round-to-nearest-even
    return (unsigned short)(u >> 16);
}
// order-preserving float -> uint mapping for max-atomics
__device__ __forceinline__ unsigned mapf(float f){
    unsigned u = __float_as_uint(f);
    return (u & 0x80000000u) ? ~u : (u | 0x80000000u);
}
__device__ __forceinline__ float unmapf(unsigned k){
    unsigned u = (k & 0x80000000u) ? (k ^ 0x80000000u) : ~k;
    return __uint_as_float(u);
}
#define KEY_NEG_INF 0x007FFFFFu               // mapf(-inf)

// ---------------------------------------------------------------------------
// Weight packing: B-fragment lane order for v_wmma_f32_16x16x32_bf16.
// frag = nb*(Kpad/32)+kb ; element (lane,j): k = 32*kb + 16*(lane>>4) + j,
// n = 16*nb + (lane&15). Rows k>=K are zero-padded (used for Kin=16).
// ---------------------------------------------------------------------------
__global__ void pack_weights(const float* __restrict__ W, unsigned short* __restrict__ dst,
                             int K, int N, int Kpad, long wStride, long dStride){
    int l = blockIdx.y;
    const float* Wl = W + (long)l * wStride;
    unsigned short* dl = dst + (long)l * dStride;
    int kblocks = Kpad >> 5;
    int total = kblocks * (N >> 4) * 512;
    for (int i = blockIdx.x * blockDim.x + threadIdx.x; i < total; i += gridDim.x * blockDim.x){
        int frag = i >> 9;
        int lane = (i >> 4) & 31;
        int j    = i & 15;
        int nb = frag / kblocks, kb = frag - nb * kblocks;
        int n = (nb << 4) + (lane & 15);
        int k = (kb << 5) + ((lane >> 4) << 4) + j;
        float v = (k < K) ? Wl[(long)k * N + n] : 0.f;
        dl[i] = f2bf(v);
    }
}

// EdgeConv W1 split: cat[xi, xj-xi]@W1 = xi@(W1a-W1b) + xj@W1b.
// Packed as one 128x256 matrix: cols [0,128)=W1a-W1b (P, dst side), [128,256)=W1b (Q, src side).
__global__ void pack_pq(const float* __restrict__ convW1, unsigned short* __restrict__ dst){
    int l = blockIdx.y;
    const float* W1 = convW1 + (long)l * 256 * 128;
    unsigned short* dl = dst + (long)l * 32768;
    const int total = 4 * 16 * 512;
    for (int i = blockIdx.x * blockDim.x + threadIdx.x; i < total; i += gridDim.x * blockDim.x){
        int frag = i >> 9;
        int lane = (i >> 4) & 31;
        int j    = i & 15;
        int nb = frag >> 2, kb = frag & 3;
        int n = (nb << 4) + (lane & 15);
        int k = (kb << 5) + ((lane >> 4) << 4) + j;
        float v;
        if (n < 128) v = W1[k * 128 + n] - W1[(128 + k) * 128 + n];
        else         v = W1[(128 + k) * 128 + (n - 128)];
        dl[i] = f2bf(v);
    }
}

// time embedding (sin/cos @ time_proj) + fused P-bias = [b1, 0] per layer
__global__ void prep_misc(const float* __restrict__ t, const float* __restrict__ tpW,
                          const float* __restrict__ tpb, const float* __restrict__ b1,
                          float* __restrict__ temb16, float* __restrict__ bias_pq){
    __shared__ float te[64];
    int tid = threadIdx.x;
    if (tid < 32){
        float fr = expf(-4.f + 8.f * (float)tid / 31.f);
        float tv = t[0] * fr;
        te[tid]      = sinf(tv);
        te[tid + 32] = cosf(tv);
    }
    __syncthreads();
    if (tid < 16){
        float acc = tpb[tid];
        for (int k = 0; k < 64; ++k) acc += te[k] * tpW[k * 16 + tid];
        temb16[tid] = acc;
    }
    for (int i = tid; i < 8 * 256; i += blockDim.x){
        int l = i >> 8, c = i & 255;
        bias_pq[i] = (c < 128) ? b1[l * 128 + c] : 0.f;
    }
}

__global__ void init_keys_k(unsigned int* __restrict__ keys, int total){
    int i = blockIdx.x * blockDim.x + threadIdx.x;
    if (i < total) keys[i] = KEY_NEG_INF;
}
__global__ void fixup_keys(unsigned int* __restrict__ keys, float* __restrict__ h, int total){
    int i = blockIdx.x * blockDim.x + threadIdx.x;
    if (i < total){ h[i] = unmapf(keys[i]); keys[i] = KEY_NEG_INF; }
}

// ---------------------------------------------------------------------------
// Node-level GEMM: out[16-row tile] = (in [+temb]) @ Wpacked + bias.
// A staged to LDS bf16; A-fragments read as two b128 LDS loads per lane.
// ---------------------------------------------------------------------------
__global__ __launch_bounds__(256) void node_gemm(
        const float* __restrict__ in, const float* __restrict__ temb,
        int Kin, int Kpad,
        const unsigned short* __restrict__ Wp,
        const float* __restrict__ bias, int Nout,
        float* __restrict__ out, int Nnodes)
{
    __shared__ __align__(16) unsigned short Abf[16 * 128];
    int row0 = blockIdx.x << 4;
    int tid = threadIdx.x;
    for (int i = tid; i < 16 * Kpad; i += 256){
        int r = i / Kpad, c = i - r * Kpad;
        int row = row0 + r;
        float v = 0.f;
        if (row < Nnodes && c < Kin){
            v = in[(long)row * Kin + c];
            if (temb) v += temb[c];
        }
        Abf[i] = f2bf(v);
    }
    __syncthreads();
    int wave = tid >> 5, lane = tid & 31;
    int lo = lane & 15, hi = lane >> 4;
    int ntiles = Nout >> 4;
    int kblocks = Kpad >> 5;
    for (int nb = wave; nb < ntiles; nb += 8){
        v8f acc = {0.f, 0.f, 0.f, 0.f, 0.f, 0.f, 0.f, 0.f};
        const unsigned short* wp = Wp + ((long)nb * kblocks) * 512 + lane * 16;
        const unsigned short* ab = &Abf[lo * Kpad + (hi << 4)];
        for (int kb = 0; kb < kblocks; ++kb){
            FragBF A, B;
            B.u[0] = *(const v4u*)(wp);
            B.u[1] = *(const v4u*)(wp + 8);
            wp += 512;
            const unsigned short* ap = ab + (kb << 5);
            A.u[0] = *(const v4u*)(ap);
            A.u[1] = *(const v4u*)(ap + 8);
            acc = __builtin_amdgcn_wmma_f32_16x16x32_bf16(false, A.v, false, B.v,
                                                          (short)0, acc, false, false);
        }
        int n = (nb << 4) + lo;
        float b = bias ? bias[n] : 0.f;
#pragma unroll
        for (int v = 0; v < 8; ++v){
            int row = row0 + v + (hi << 3);
            if (row < Nnodes) out[(long)row * Nout + n] = acc[v] + b;
        }
    }
}

// ---------------------------------------------------------------------------
// Edge stage: workgroup = 16 tiles of 16 edges (256 edges). W2 staged in LDS
// once per WG; each wave owns 2 tiles so every B fragment feeds 2 WMMAs
// (independent accumulators -> XDL pipelining, halved B traffic).
// hidden = relu(P[dst]+Q[src]) -> LDS bf16; hidden @ W2 + b2; keyed
// atomic-max scatter into dst rows.
// ---------------------------------------------------------------------------
__global__ __launch_bounds__(256) void edge_conv(
        const float* __restrict__ PQ, const int* __restrict__ src, const int* __restrict__ dst,
        int E, int N,
        const unsigned short* __restrict__ W2p, const float* __restrict__ b2,
        unsigned int* __restrict__ keys)
{
    __shared__ __align__(16) unsigned short W2s[32 * 512];     // 32 KB packed W2
    __shared__ __align__(16) unsigned short Hbf[16][16 * 128]; // 64 KB hidden tiles
    __shared__ int Dsh[16][16];
    int tid = threadIdx.x, wave = tid >> 5, lane = tid & 31;
    int Etot = E + N;
    int tileBase = blockIdx.x * 16;

    // stage all W2 fragments to LDS (cooperative, 16B per thread per iter)
    {
        const v4u* gsrc = (const v4u*)W2p;
        v4u* ldst = (v4u*)W2s;
        for (int i = tid; i < 2048; i += 256) ldst[i] = gsrc[i];
    }

    // hidden tiles: wave owns local tiles (wave) and (wave+8)
    for (int tt = 0; tt < 2; ++tt){
        int tl = wave + (tt << 3);
        int e0 = (tileBase + tl) << 4;
        unsigned short* Hw = &Hbf[tl][0];
        for (int el = 0; el < 16; ++el){
            int e = e0 + el;
            int s, d;
            if (e < E){ s = src[e]; d = dst[e]; }
            else { int nn = e - E; if (nn >= N) nn = 0; s = nn; d = nn; } // self-loops / OOB clamp
            if (lane == 0) Dsh[tl][el] = d;
            const float4* Pr = (const float4*)(PQ + (long)d * 256);       // b1 folded into P
            const float4* Qr = (const float4*)(PQ + (long)s * 256 + 128);
            float4 p = Pr[lane];
            float4 q = Qr[lane];
            unsigned long long pk =
                  (unsigned long long)f2bf(fmaxf(p.x + q.x, 0.f))
                | ((unsigned long long)f2bf(fmaxf(p.y + q.y, 0.f)) << 16)
                | ((unsigned long long)f2bf(fmaxf(p.z + q.z, 0.f)) << 32)
                | ((unsigned long long)f2bf(fmaxf(p.w + q.w, 0.f)) << 48);
            *(unsigned long long*)&Hw[(el << 7) + (lane << 2)] = pk;
        }
    }
    __syncthreads();

    int lo = lane & 15, hi = lane >> 4;
    int tl0 = wave, tl1 = wave + 8;
    FragBF A0[4], A1[4];
    const unsigned short* ab0 = &Hbf[tl0][(lo << 7) + (hi << 4)];
    const unsigned short* ab1 = &Hbf[tl1][(lo << 7) + (hi << 4)];
#pragma unroll
    for (int kb = 0; kb < 4; ++kb){
        A0[kb].u[0] = *(const v4u*)(ab0 + (kb << 5));
        A0[kb].u[1] = *(const v4u*)(ab0 + (kb << 5) + 8);
        A1[kb].u[0] = *(const v4u*)(ab1 + (kb << 5));
        A1[kb].u[1] = *(const v4u*)(ab1 + (kb << 5) + 8);
    }
    int e0a = (tileBase + tl0) << 4;
    int e0b = (tileBase + tl1) << 4;
    for (int nb = 0; nb < 8; ++nb){
        v8f acc0 = {0.f, 0.f, 0.f, 0.f, 0.f, 0.f, 0.f, 0.f};
        v8f acc1 = {0.f, 0.f, 0.f, 0.f, 0.f, 0.f, 0.f, 0.f};
        const unsigned short* wp = &W2s[(nb * 4) * 512 + lane * 16];
#pragma unroll
        for (int kb = 0; kb < 4; ++kb){
            FragBF B;
            B.u[0] = *(const v4u*)(wp);
            B.u[1] = *(const v4u*)(wp + 8);
            wp += 512;
            acc0 = __builtin_amdgcn_wmma_f32_16x16x32_bf16(false, A0[kb].v, false, B.v,
                                                           (short)0, acc0, false, false);
            acc1 = __builtin_amdgcn_wmma_f32_16x16x32_bf16(false, A1[kb].v, false, B.v,
                                                           (short)0, acc1, false, false);
        }
        int n = (nb << 4) + lo;
        float bb = b2[n];
#pragma unroll
        for (int v = 0; v < 8; ++v){
            int m = v + (hi << 3);
            int e = e0a + m;
            if (e < Etot)
                atomicMax(&keys[(long)Dsh[tl0][m] * 128 + n], mapf(acc0[v] + bb));
            e = e0b + m;
            if (e < Etot)
                atomicMax(&keys[(long)Dsh[tl1][m] * 128 + n], mapf(acc1[v] + bb));
        }
    }
}

// ---------------------------------------------------------------------------
extern "C" void kernel_launch(void* const* d_in, const int* in_sizes, int n_in,
                              void* d_out, int out_size, void* d_ws, size_t ws_size,
                              hipStream_t stream)
{
    (void)n_in; (void)out_size; (void)ws_size;
    const float* x     = (const float*)d_in[0];
    const int*   eidx  = (const int*)d_in[1];
    const float* t     = (const float*)d_in[2];
    const float* nembW = (const float*)d_in[3];
    const float* nembB = (const float*)d_in[4];
    const float* cW1   = (const float*)d_in[5];
    const float* cB1   = (const float*)d_in[6];
    const float* cW2   = (const float*)d_in[7];
    const float* cB2   = (const float*)d_in[8];
    const float* encW  = (const float*)d_in[9];
    const float* encB  = (const float*)d_in[10];
    const float* decW  = (const float*)d_in[11];
    const float* decB  = (const float*)d_in[12];
    const float* tpW   = (const float*)d_in[13];
    const float* tpB   = (const float*)d_in[14];

    int N = in_sizes[0] / 16;
    int E = in_sizes[1] / 2;
    const int* srcp = eidx;
    const int* dstp = eidx + E;

    char* ws = (char*)d_ws;
    size_t off = 0;
    auto take = [&](size_t bytes) -> char* {
        off = (off + 255) & ~(size_t)255;
        char* p = ws + off;
        off += bytes;
        return p;
    };
    float*          temb16  = (float*)take(16 * 4);
    float*          bias_pq = (float*)take(8 * 256 * 4);
    unsigned short* pknode  = (unsigned short*)take(8L   * 512 * 2);   // 16(->32 pad) x 128
    unsigned short* pkenc   = (unsigned short*)take(32L  * 512 * 2);   // 128 x 128
    unsigned short* pkdec   = (unsigned short*)take(4L   * 512 * 2);   // 128 x 16
    unsigned short* pkpq    = (unsigned short*)take(8L * 64 * 512 * 2);// per layer 128 x 256
    unsigned short* pkw2    = (unsigned short*)take(8L * 32 * 512 * 2);// per layer 128 x 128
    float*          h       = (float*)take((size_t)N * 128 * 4);
    float*          PQ      = (float*)take((size_t)N * 256 * 4);
    unsigned int*   keys    = (unsigned int*)take((size_t)N * 128 * 4);

    prep_misc   <<<1, 256, 0, stream>>>(t, tpW, tpB, cB1, temb16, bias_pq);
    pack_weights<<<dim3(16, 1), 256, 0, stream>>>(nembW, pknode, 16, 128, 32, 0, 0);
    pack_weights<<<dim3(64, 1), 256, 0, stream>>>(encW, pkenc, 128, 128, 128, 0, 0);
    pack_weights<<<dim3(8, 1),  256, 0, stream>>>(decW, pkdec, 128, 16, 128, 0, 0);
    pack_weights<<<dim3(64, 8), 256, 0, stream>>>(cW2, pkw2, 128, 128, 128, 128L * 128, 32768);
    pack_pq     <<<dim3(128, 8), 256, 0, stream>>>(cW1, pkpq);
    init_keys_k <<<(N * 128 + 255) / 256, 256, 0, stream>>>(keys, N * 128);

    int rowTiles = (N + 15) / 16;
    int tiles16  = (E + N + 15) / 16;
    int edgeWGs  = (tiles16 + 15) / 16;

    // h = (x + temb) @ node_emb_W + b
    node_gemm<<<rowTiles, 256, 0, stream>>>(x, temb16, 16, 32, pknode, nembB, 128, h, N);

    for (int l = 0; l < 8; ++l){
        if (l == 4)  // encoder fc between layer 3 and 4 (in-place: each WG reads then writes its own rows)
            node_gemm<<<rowTiles, 256, 0, stream>>>(h, nullptr, 128, 128, pkenc, encB, 128, h, N);
        node_gemm<<<rowTiles, 256, 0, stream>>>(h, nullptr, 128, 128,
                                                pkpq + (size_t)l * 32768,
                                                bias_pq + l * 256, 256, PQ, N);
        edge_conv<<<edgeWGs, 256, 0, stream>>>(PQ, srcp, dstp, E, N,
                                               pkw2 + (size_t)l * 16384, cB2 + l * 128, keys);
        fixup_keys<<<(N * 128 + 255) / 256, 256, 0, stream>>>(keys, h, N * 128);
    }
    node_gemm<<<rowTiles, 256, 0, stream>>>(h, nullptr, 128, 128, pkdec, decB, 16,
                                            (float*)d_out, N);
}
